// MomentumSMoELayer_9320079032503
// MI455X (gfx1250) — matrix operand order repaired and checked by reading, not compiled, not run on hardware
//
#include <hip/hip_runtime.h>
#include <hip/hip_bf16.h>
#include <math.h>

// ---------------------------------------------------------------------------
// MomentumSMoE for gfx1250 (MI455X): gathered top-2 expert tiles, bf16 WMMA,
// async global->LDS staging, compact slot scatter + combine/LayerNorm pass.
// B=4 S=2048 D=1024 H=2048 E=8 K=2
// ---------------------------------------------------------------------------

#define BB 4
#define SS 2048
#define DD 1024
#define HH 2048
#define EE 8
#define NTOK (BB * SS)              // 8192 tokens
#define NASSIGN (NTOK * 2)          // 16384 (token,expert) assignments
#define MT 16                       // rows per WMMA tile
#define MAXTILES 1040               // >= 16384/16 + 7
#define MU_C 0.7f
#define GAMMA_C 1.0f
#define LN_EPS_C 1e-5f

typedef __attribute__((ext_vector_type(16))) __bf16 v16bf;
typedef __attribute__((ext_vector_type(8)))  float  v8f;
typedef int v4i __attribute__((ext_vector_type(4)));
typedef __attribute__((address_space(1))) v4i* g4ptr;   // global int4*
typedef __attribute__((address_space(3))) v4i* l4ptr;   // LDS int4*

// A-fragment (16x32 bf16, lane=M): 16 contiguous K values (32 bytes)
__device__ __forceinline__ v16bf load_fragA(const __bf16* p) {
    union { v16bf v; uint4 q[2]; } u;
    u.q[0] = *(const uint4*)(p);
    u.q[1] = *(const uint4*)(p + 8);
    return u.v;
}
// B-fragment (32x16 bf16, lane=N): 16 contiguous K values (32 bytes)
__device__ __forceinline__ v16bf load_fragB(const __bf16* p) {
    union { v16bf v; uint4 q[2]; } u;
    u.q[0] = *(const uint4*)(p);
    u.q[1] = *(const uint4*)(p + 8);
    return u.v;
}
// gfx1250 async DMA: 16B global -> LDS, tracked by ASYNCcnt
__device__ __forceinline__ void async_copy_b128(const __bf16* g, __bf16* l) {
#if __has_builtin(__builtin_amdgcn_global_load_async_to_lds_b128)
    __builtin_amdgcn_global_load_async_to_lds_b128((g4ptr)g, (l4ptr)l, 0, 0);
#else
    *(uint4*)l = *(const uint4*)g;
#endif
}
__device__ __forceinline__ void wait_async0() {
#if __has_builtin(__builtin_amdgcn_s_wait_asynccnt)
    __builtin_amdgcn_s_wait_asynccnt(0);
#elif __has_builtin(__builtin_amdgcn_global_load_async_to_lds_b128)
    asm volatile("s_wait_asynccnt 0x0" ::: "memory");
#endif
}

// ---------------------------------------------------------------------------
// meta[] ints in ws: [0..7]=cnt  [8..15]=cnt2  [16..23]=base  [24]=ntiles
// ---------------------------------------------------------------------------
__global__ void zero_meta(int* meta) {
    if (threadIdx.x < 64) meta[threadIdx.x] = 0;
}

// x f32 -> xb bf16 (4 elems/thread)
__global__ void convert_xb(const float* __restrict__ x, __bf16* __restrict__ xb) {
    const size_t i = (size_t)blockIdx.x * blockDim.x + threadIdx.x;
    float4 v = ((const float4*)x)[i];
    union { __bf16 h[4]; uint2 u; } p;
    p.h[0] = (__bf16)v.x; p.h[1] = (__bf16)v.y;
    p.h[2] = (__bf16)v.z; p.h[3] = (__bf16)v.w;
    ((uint2*)xb)[i] = p.u;
}

// tiled transpose + convert: in [E][R][C] f32 -> out [E][C][R] bf16
__global__ void transpose_to_bf16(const float* __restrict__ in,
                                  __bf16* __restrict__ out, int R, int C) {
    __shared__ float tile[32][33];
    const int e  = blockIdx.z;
    const int c0 = blockIdx.x * 32;
    const int r0 = blockIdx.y * 32;
    const float* ip = in  + (size_t)e * R * C;
    __bf16*      op = out + (size_t)e * R * C;
    const int tx = threadIdx.x, ty = threadIdx.y;
#pragma unroll
    for (int j = 0; j < 32; j += 8)
        tile[ty + j][tx] = ip[(size_t)(r0 + ty + j) * C + (c0 + tx)];
    __syncthreads();
#pragma unroll
    for (int j = 0; j < 32; j += 8)
        op[(size_t)(c0 + ty + j) * R + (r0 + tx)] = (__bf16)tile[tx][ty + j];
}

// ---------------------------------------------------------------------------
// Router: one wave per token -> top-2 experts + softmax weights
// ---------------------------------------------------------------------------
__global__ void gate_kernel(const float* __restrict__ x,
                            const float* __restrict__ Wg,
                            const float* __restrict__ bg,
                            int2* __restrict__ tokexp,
                            float2* __restrict__ tokw) {
    const int lane = threadIdx.x & 31;
    const int wv   = threadIdx.x >> 5;
    const int tok  = blockIdx.x * 8 + wv;
    const float* xp = x + (size_t)tok * DD;

    float acc[EE];
#pragma unroll
    for (int e = 0; e < EE; e++) acc[e] = 0.f;
    for (int d = lane; d < DD; d += 32) {
        float xv = xp[d];
        const float4* wr = (const float4*)(Wg + (size_t)d * EE);
        float4 a = wr[0], b = wr[1];
        acc[0] += xv * a.x; acc[1] += xv * a.y;
        acc[2] += xv * a.z; acc[3] += xv * a.w;
        acc[4] += xv * b.x; acc[5] += xv * b.y;
        acc[6] += xv * b.z; acc[7] += xv * b.w;
    }
#pragma unroll
    for (int e = 0; e < EE; e++)
        for (int off = 16; off; off >>= 1)
            acc[e] += __shfl_down(acc[e], off, 32);

    if (lane == 0) {
        float l[EE];
#pragma unroll
        for (int e = 0; e < EE; e++) l[e] = acc[e] + bg[e];
        int   e0 = 0, e1 = -1;
        float v0 = l[0], v1 = -3.0e38f;
#pragma unroll
        for (int e = 1; e < EE; e++) {
            if (l[e] > v0)      { v1 = v0; e1 = e0; v0 = l[e]; e0 = e; }
            else if (l[e] > v1) { v1 = l[e]; e1 = e; }
        }
        float w1v = __expf(v1 - v0);
        float inv = 1.f / (1.f + w1v);
        tokexp[tok] = make_int2(e0, e1);
        tokw[tok]   = make_float2(inv, w1v * inv);
    }
}

__global__ void count_kernel(const int2* __restrict__ tokexp, int* meta) {
    const int tok = blockIdx.x * blockDim.x + threadIdx.x;
    if (tok < NTOK) {
        int2 e2 = tokexp[tok];
        atomicAdd(&meta[e2.x], 1);
        atomicAdd(&meta[e2.y], 1);
    }
}

// single-thread: prefix sums + tile descriptors {expert, slotbase, nvalid}
__global__ void scheduler_kernel(int* meta, int4* tdesc) {
    if (threadIdx.x == 0 && blockIdx.x == 0) {
        int base = 0, nt = 0;
        for (int e = 0; e < EE; e++) {
            meta[16 + e] = base;
            int c  = meta[e];
            int t0 = (c + MT - 1) / MT;
            for (int i = 0; i < t0 && nt < MAXTILES; i++) {
                int nv = c - i * MT; if (nv > MT) nv = MT;
                tdesc[nt] = make_int4(e, base + i * MT, nv, 0);
                nt++;
            }
            base += c;
        }
        meta[24] = nt;
    }
}

__global__ void assign_kernel(const int2* __restrict__ tokexp,
                              const float2* __restrict__ tokw,
                              int* meta, int* __restrict__ list,
                              float* __restrict__ wgt,
                              int2* __restrict__ slotIdx) {
    const int tok = blockIdx.x * blockDim.x + threadIdx.x;
    if (tok >= NTOK) return;
    int2   e2 = tokexp[tok];
    float2 w2 = tokw[tok];
    int p0 = atomicAdd(&meta[8 + e2.x], 1);
    int g0 = meta[16 + e2.x] + p0;
    list[g0] = tok; wgt[g0] = w2.x;
    int p1 = atomicAdd(&meta[8 + e2.y], 1);
    int g1 = meta[16 + e2.y] + p1;
    list[g1] = tok; wgt[g1] = w2.y;
    slotIdx[tok] = make_int2(g0, g1);
}

// ---------------------------------------------------------------------------
// Per-tile FFN: 16 gathered tokens, one expert.  8 waves; wave w owns output
// columns [w*128, w*128+128).  Writes gate-weighted FFN rows to slotout.
// ---------------------------------------------------------------------------
__global__ __launch_bounds__(256) void ffn_tile(
    const int* __restrict__ meta, const int4* __restrict__ tdesc,
    const int* __restrict__ list, const float* __restrict__ wgt,
    const __bf16* __restrict__ xb, const float* __restrict__ b1,
    const __bf16* __restrict__ W1T,   // [E][H][D]
    const __bf16* __restrict__ W2T,   // [E][D][H]
    float* __restrict__ slotout) {

    __shared__ __bf16 sX[MT * DD];     // 32 KB gathered x rows (bf16)
    __shared__ __bf16 sH[MT * 256];    //  8 KB current h super-chunk
    __shared__ int    sTok[MT];
    __shared__ float  sGw[MT];

    if (blockIdx.x >= meta[24]) return;
    const int4 td   = tdesc[blockIdx.x];
    const int e     = td.x;
    const int gbase = td.y;
    const int nv    = td.z;

    const int tid  = threadIdx.x;
    const int lane = tid & 31;
    const int wv   = tid >> 5;
    const int nlo  = lane & 15;
    const int hiK  = lane >> 4;

    if (tid < MT) {
        const bool v = tid < nv;
        sTok[tid] = list[v ? (gbase + tid) : gbase];
        sGw[tid]  = v ? wgt[gbase + tid] : 0.f;
    }
    __syncthreads();

    // async DMA: gather 16 token rows (2 KB each) into LDS
    for (int c = tid; c < MT * DD / 8; c += 256) {   // 16B chunks
        const int m = c >> 7, off = (c & 127) * 8;
        async_copy_b128(xb + (size_t)sTok[m] * DD + off, &sX[m * DD + off]);
    }
    wait_async0();
    __syncthreads();

    const __bf16* W1e = W1T + (size_t)e * HH * DD;
    const __bf16* W2e = W2T + (size_t)e * DD * HH;

    const v8f zero = {0.f, 0.f, 0.f, 0.f, 0.f, 0.f, 0.f, 0.f};
    v8f acc[8];
#pragma unroll
    for (int t = 0; t < 8; t++) acc[t] = zero;

    for (int sc = 0; sc < HH / 256; sc++) {
        const int hbase = sc * 256 + wv * 32;
        v8f c0 = zero, c1 = zero;

        // layer 1: h[16 x 32] = X[16 x 1024] * W1[:, hbase..hbase+31]
        const __bf16* bcol0 = W1e + (size_t)(hbase + nlo) * DD + hiK * 16;
        const __bf16* bcol1 = bcol0 + (size_t)16 * DD;
#pragma unroll 4
        for (int kk = 0; kk < DD / 32; kk++) {
            const int k0 = kk * 32;
            v16bf A  = load_fragA(&sX[nlo * DD + k0 + hiK * 16]);
            v16bf B0 = load_fragB(bcol0 + k0);
            v16bf B1 = load_fragB(bcol1 + k0);
            c0 = __builtin_amdgcn_wmma_f32_16x16x32_bf16(
                    false, A, false, B0, (short)0, c0, false, false);
            c1 = __builtin_amdgcn_wmma_f32_16x16x32_bf16(
                    false, A, false, B1, (short)0, c1, false, false);
        }
        const float b1v0 = b1[(size_t)e * HH + hbase + nlo];
        const float b1v1 = b1[(size_t)e * HH + hbase + 16 + nlo];
#pragma unroll
        for (int r = 0; r < 8; r++) {
            const int   m = r + hiK * 8;
            const float g = sGw[m];
            float h0 = fmaxf(c0[r] + b1v0, 0.f) * g;
            float h1 = fmaxf(c1[r] + b1v1, 0.f) * g;
            sH[m * 256 + wv * 32 + nlo]      = (__bf16)h0;
            sH[m * 256 + wv * 32 + 16 + nlo] = (__bf16)h1;
        }
        __syncthreads();

        // layer 2: acc[16 x 128] += h[16 x 256] * W2[hchunk, my cols]
#pragma unroll
        for (int kk = 0; kk < 8; kk++) {
            v16bf A = load_fragA(&sH[nlo * 256 + kk * 32 + hiK * 16]);
            const int kglob = sc * 256 + kk * 32 + hiK * 16;
#pragma unroll
            for (int t = 0; t < 8; t++) {
                const __bf16* bp =
                    W2e + (size_t)(wv * 128 + t * 16 + nlo) * HH + kglob;
                v16bf B = load_fragB(bp);
                acc[t] = __builtin_amdgcn_wmma_f32_16x16x32_bf16(
                            false, A, false, B, (short)0, acc[t], false, false);
            }
        }
        __syncthreads();
    }

    // scatter valid rows to compact per-assignment slots
#pragma unroll
    for (int t = 0; t < 8; t++) {
        const int col = wv * 128 + t * 16 + nlo;
#pragma unroll
        for (int r = 0; r < 8; r++) {
            const int m = r + hiK * 8;
            if (m < nv)
                slotout[(size_t)(gbase + m) * DD + col] = acc[t][r];
        }
    }
}

// ---------------------------------------------------------------------------
// Combine two expert slots + gate-weighted b2, momentum, residual, LayerNorm.
// One wave per token.
// ---------------------------------------------------------------------------
__global__ __launch_bounds__(256) void combine_ln(
    const float* __restrict__ x, const float* __restrict__ mom,
    const float* __restrict__ b2, const float* __restrict__ lng,
    const float* __restrict__ lnb, const int2* __restrict__ tokexp,
    const float2* __restrict__ tokw, const int2* __restrict__ slotIdx,
    const float* __restrict__ slotout,
    float* __restrict__ out0, float* __restrict__ out1) {

    const int lane = threadIdx.x & 31;
    const int wv   = threadIdx.x >> 5;
    const int tok  = blockIdx.x * 8 + wv;

    const int2   e2 = tokexp[tok];
    const float2 w2 = tokw[tok];
    const int2   s2 = slotIdx[tok];
    const float* so0 = slotout + (size_t)s2.x * DD;
    const float* so1 = slotout + (size_t)s2.y * DD;
    const float* b20 = b2 + (size_t)e2.x * DD;
    const float* b21 = b2 + (size_t)e2.y * DD;

    float pre[DD / 32];
    float sum = 0.f, sq = 0.f;
#pragma unroll
    for (int i = 0; i < DD / 32; i++) {
        const int d  = lane + i * 32;
        float eo = so0[d] + so1[d] + w2.x * b20[d] + w2.y * b21[d];
        float mn = MU_C * mom[(size_t)tok * DD + d] - eo;
        float p  = x[(size_t)tok * DD + d] + GAMMA_C * mn;
        out1[(size_t)tok * DD + d] = mn;
        pre[i] = p;
        sum += p; sq += p * p;
    }
    for (int off = 16; off; off >>= 1) {
        sum += __shfl_xor(sum, off, 32);
        sq  += __shfl_xor(sq,  off, 32);
    }
    const float mean = sum * (1.f / DD);
    const float var  = sq * (1.f / DD) - mean * mean;
    const float rstd = rsqrtf(var + LN_EPS_C);
#pragma unroll
    for (int i = 0; i < DD / 32; i++) {
        const int d = lane + i * 32;
        out0[(size_t)tok * DD + d] = (pre[i] - mean) * rstd * lng[d] + lnb[d];
    }
}

// ---------------------------------------------------------------------------
extern "C" void kernel_launch(void* const* d_in, const int* in_sizes, int n_in,
                              void* d_out, int out_size, void* d_ws, size_t ws_size,
                              hipStream_t stream) {
    const float* x   = (const float*)d_in[0];
    const float* mom = (const float*)d_in[1];
    const float* Wg  = (const float*)d_in[2];
    const float* bg  = (const float*)d_in[3];
    const float* W1  = (const float*)d_in[4];
    const float* b1  = (const float*)d_in[5];
    const float* W2  = (const float*)d_in[6];
    const float* b2  = (const float*)d_in[7];
    const float* lng = (const float*)d_in[8];
    const float* lnb = (const float*)d_in[9];

    float* out0 = (float*)d_out;
    float* out1 = out0 + (size_t)NTOK * DD;

    // ---- workspace layout ----------------------------------------------------
    char* ws = (char*)d_ws;
    const size_t OFF_META    = 0;                                  // 256 B
    const size_t OFF_TOKEXP  = 256;                                // 64 KB
    const size_t OFF_TOKW    = OFF_TOKEXP  + (size_t)NTOK * 8;
    const size_t OFF_SLOTIDX = OFF_TOKW    + (size_t)NTOK * 8;
    const size_t OFF_LIST    = OFF_SLOTIDX + (size_t)NTOK * 8;
    const size_t OFF_WGT     = OFF_LIST    + (size_t)NASSIGN * 4;
    const size_t OFF_TDESC   = OFF_WGT     + (size_t)NASSIGN * 4;
    const size_t OFF_XB      = OFF_TDESC   + (size_t)MAXTILES * 16 + 256;
    const size_t OFF_W1T     = OFF_XB      + (size_t)NTOK * DD * 2;
    const size_t OFF_W2T     = OFF_W1T     + (size_t)EE * DD * HH * 2;
    const size_t OFF_SLOTOUT = OFF_W2T     + (size_t)EE * DD * HH * 2;

    int*    meta    = (int*)   (ws + OFF_META);
    int2*   tokexp  = (int2*)  (ws + OFF_TOKEXP);
    float2* tokw    = (float2*)(ws + OFF_TOKW);
    int2*   slotIdx = (int2*)  (ws + OFF_SLOTIDX);
    int*    list    = (int*)   (ws + OFF_LIST);
    float*  wgt     = (float*) (ws + OFF_WGT);
    int4*   tdesc   = (int4*)  (ws + OFF_TDESC);
    __bf16* xb      = (__bf16*)(ws + OFF_XB);
    __bf16* W1T     = (__bf16*)(ws + OFF_W1T);
    __bf16* W2T     = (__bf16*)(ws + OFF_W2T);
    float*  slotout = (float*) (ws + OFF_SLOTOUT);

    // ---- pipeline ------------------------------------------------------------
    zero_meta<<<1, 64, 0, stream>>>(meta);
    convert_xb<<<(NTOK * DD / 4) / 256, 256, 0, stream>>>(x, xb);
    // W1 [E][D][H] -> W1T [E][H][D] ; W2 [E][H][D] -> W2T [E][D][H]
    transpose_to_bf16<<<dim3(HH / 32, DD / 32, EE), dim3(32, 8), 0, stream>>>(
        W1, W1T, DD, HH);
    transpose_to_bf16<<<dim3(DD / 32, HH / 32, EE), dim3(32, 8), 0, stream>>>(
        W2, W2T, HH, DD);
    gate_kernel<<<NTOK / 8, 256, 0, stream>>>(x, Wg, bg, tokexp, tokw);
    count_kernel<<<NTOK / 256, 256, 0, stream>>>(tokexp, meta);
    scheduler_kernel<<<1, 32, 0, stream>>>(meta, tdesc);
    assign_kernel<<<NTOK / 256, 256, 0, stream>>>(tokexp, tokw, meta,
                                                  list, wgt, slotIdx);
    ffn_tile<<<MAXTILES, 256, 0, stream>>>(meta, tdesc, list, wgt, xb, b1,
                                           W1T, W2T, slotout);
    combine_ln<<<NTOK / 8, 256, 0, stream>>>(x, mom, b2, lng, lnb, tokexp,
                                             tokw, slotIdx, slotout,
                                             out0, out1);
}
